// BMN_model_48507360641153
// MI455X (gfx1250) — matrix operand order repaired and checked by reading, not compiled, not run on hardware
//
#include <hip/hip_runtime.h>
#include <hip/hip_bf16.h>

// ---------------------------------------------------------------------------
// BMN forward for MI455X (gfx1250, wave32, WMMA).
// Heavy path: double-buffered v_wmma_f32_16x16x32_f16 tiled GEMM.
//   - A tile prefetched via global_load_async_to_lds_b128 (ASYNCcnt), no VGPRs
//   - B tile via coalesced b128 global loads + LDS transpose scatter
//   - copies for step s+1 overlap the 8 WMMAs of step s; s_wait_asynccnt 0
//     + workgroup barrier close each step.
// f16 operands / f32 accumulate (mask in [0,1], activations O(1)).
// ---------------------------------------------------------------------------

typedef _Float16 v16h __attribute__((ext_vector_type(16)));
typedef float    v8f  __attribute__((ext_vector_type(8)));

#define T100 100
#define POS  10000   // T*T

// ---------------- small VALU kernels (cheap parts of the net) ----------------

__global__ void conv1d_act(const float* __restrict__ in, const float* __restrict__ w,
                           const float* __restrict__ bias, float* __restrict__ out,
                           int Bsz, int IC, int OC, int K, int pad, int groups, int act)
{
    int idx = blockIdx.x * blockDim.x + threadIdx.x;
    int total = Bsz * OC * T100;
    if (idx >= total) return;
    int t  = idx % T100;
    int oc = (idx / T100) % OC;
    int b  = idx / (T100 * OC);
    int icpg = IC / groups, ocpg = OC / groups;
    int g = oc / ocpg;
    const float* wrow = w + (size_t)oc * icpg * K;
    const float* inb  = in + ((size_t)b * IC + (size_t)g * icpg) * T100;
    float acc = bias[oc];
    for (int ic = 0; ic < icpg; ++ic)
        for (int k = 0; k < K; ++k) {
            int tt = t + k - pad;
            if (tt >= 0 && tt < T100)
                acc += wrow[ic * K + k] * inb[(size_t)ic * T100 + tt];
        }
    if (act == 1)      acc = fmaxf(acc, 0.f);
    else if (act == 2) acc = 1.f / (1.f + __expf(-acc));
    out[idx] = acc;
}

// s[b][o][n*100+t] = sum_c w3d[o][c][n] * p[b][c][t]  -> f16 (GEMM A operand)
__global__ void s_contract(const float* __restrict__ p, const float* __restrict__ w3d,
                           _Float16* __restrict__ s)
{
    int idx = blockIdx.x * blockDim.x + threadIdx.x;
    const int total = 4 * 512 * 3200;
    if (idx >= total) return;
    int nt = idx % 3200;
    int o  = (idx / 3200) % 512;
    int b  = idx / (3200 * 512);
    int n = nt / T100, t = nt % T100;
    const float* pb = p + ((size_t)b * 128) * T100 + t;
    const float* wv = w3d + (size_t)o * 128 * 32 + n;
    float acc = 0.f;
    for (int c = 0; c < 128; ++c)
        acc += wv[(size_t)c * 32] * pb[(size_t)c * T100];
    s[idx] = (_Float16)acc;
}

// maskT[(n*100+t)][col] = (f16) sample_mask[t][n*10000+col]
__global__ void mask_repack(const float* __restrict__ mask, _Float16* __restrict__ mT)
{
    int idx = blockIdx.x * blockDim.x + threadIdx.x;
    const int total = 3200 * POS;
    if (idx >= total) return;
    int col = idx % POS;
    int nt  = idx / POS;
    int n = nt / T100, t = nt % T100;
    mT[(size_t)idx] = (_Float16)mask[(size_t)t * (32 * POS) + (size_t)n * POS + col];
}

// plain f32 -> f16 repack (for GEMM A weight operands)
__global__ void cvt_f16(const float* __restrict__ in, _Float16* __restrict__ out, int n)
{
    int i = blockIdx.x * blockDim.x + threadIdx.x;
    if (i < n) out[i] = (_Float16)in[i];
}

// im2col for 3x3 pad=1 conv on [128,100,100]: col[ic*9+dy*3+dx][y*100+x]
__global__ void im2col3x3(const float* __restrict__ in, _Float16* __restrict__ col)
{
    int idx = blockIdx.x * blockDim.x + threadIdx.x;
    const int total = 1152 * POS;
    if (idx >= total) return;
    int pos = idx % POS;
    int k   = idx / POS;
    int ic = k / 9;
    int r  = k - ic * 9;
    int dy = r / 3, dx = r % 3;
    int y = pos / T100, x = pos % T100;
    int yy = y + dy - 1, xx = x + dx - 1;
    float v = 0.f;
    if (yy >= 0 && yy < T100 && xx >= 0 && xx < T100)
        v = in[(size_t)ic * POS + yy * T100 + xx];
    col[(size_t)idx] = (_Float16)v;
}

// final 1x1 conv 128->2 + sigmoid (M=2, not worth WMMA)
__global__ void conv1x1_sig(const float* __restrict__ x, const float* __restrict__ w,
                            const float* __restrict__ bias, float* __restrict__ out)
{
    int idx = blockIdx.x * blockDim.x + threadIdx.x;
    const int total = 4 * 2 * POS;
    if (idx >= total) return;
    int pos = idx % POS;
    int o   = (idx / POS) % 2;
    int b   = idx / (2 * POS);
    const float* xb = x + (size_t)b * 128 * POS + pos;
    const float* wr = w + o * 128;
    float acc = bias[o];
    for (int c = 0; c < 128; ++c)
        acc += wr[c] * xb[(size_t)c * POS];
    out[idx] = 1.f / (1.f + __expf(-acc));
}

// ---------------- CDNA5 async copy: global -> LDS, 16B per lane --------------

__device__ __forceinline__ void async_copy16(unsigned lds_off, const _Float16* g)
{
    asm volatile("global_load_async_to_lds_b128 %0, %1, off"
                 :: "v"(lds_off), "v"((unsigned long long)g)
                 : "memory");
}
__device__ __forceinline__ void wait_async0()
{
    asm volatile("s_wait_asynccnt 0x0" ::: "memory");
}

// ---------------- WMMA GEMM: C = act(A x B + bias), A/B f16, C = CT ----------
// A: [M x K] f16 row-major (lda); B: [K x N] f16 row-major (ldb); bias[M].
// M % 64 == 0, K % 32 == 0, N % 8 == 0 at all call sites; N-edge guarded.
// Block = 128 threads (4 waves, 2x2); each wave owns 32x64 = 2x4 WMMA tiles.
// Double-buffered LDS: step s+1 tiles are fetched (async for A) while the
// WMMAs of step s execute.

template <typename CT>
__global__ __launch_bounds__(128)
void gemm_wmma(const _Float16* __restrict__ A, long long strideA, int lda,
               const _Float16* __restrict__ Bm, long long strideB, int ldb,
               CT* __restrict__ C, long long strideC, int ldc,
               const float* __restrict__ bias, int N, int Kd, int act)
{
    constexpr int BM = 64, BN = 128, BK = 32, LDS_K = BK + 8;  // 80B rows
    __shared__ _Float16 As[2][BM][LDS_K];   // row-major [m][k]
    __shared__ _Float16 Bs[2][BN][LDS_K];   // transposed [n][k] -> K contiguous

    const int zb = blockIdx.z;
    const _Float16* Ab = A  + (size_t)zb * strideA;
    const _Float16* Bb = Bm + (size_t)zb * strideB;
    CT*             Cb = C  + (size_t)zb * strideC;

    const int mBlock = blockIdx.y * BM;
    const int nBlock = blockIdx.x * BN;
    const int tid   = threadIdx.x;
    const int wave  = tid >> 5, lane = tid & 31;
    const int lhalf = lane >> 4, lmod = lane & 15;
    const int wr = wave >> 1, wc = wave & 1;

    v8f acc[2][4] = {};

    // ---- per-thread staging coordinates (constant over K-steps) ----
    // A: chunks c = tid and tid+128 -> rows aRow0 / aRow0+32, same K-segment.
    const int aRow0 = tid >> 2, aSeg = (tid & 3) * 8;        // 8-half chunk
    const _Float16* aSrc0 = Ab + (size_t)(mBlock + aRow0) * lda + aSeg;
    const _Float16* aSrc1 = aSrc0 + (size_t)32 * lda;
    // B: 4 chunks, krow = tid>>4 + cc*8, nseg constant, valid constant.
    const int bK0   = tid >> 4;                 // 0..7
    const int nseg  = (tid & 15) * 8;           // 0..120
    const int col   = nBlock + nseg;
    const bool bValid = (col < N);              // N%8==0 -> chunk all-or-nothing
    const _Float16* bSrc[4];
#pragma unroll
    for (int cc = 0; cc < 4; ++cc)
        bSrc[cc] = Bb + (size_t)(bK0 + cc * 8) * ldb + col;

    // stage tile for K-offset currently addressed by aSrc*/bSrc into buffer `buf`
    auto stage = [&](int buf) {
        async_copy16((unsigned)(unsigned long long)&As[buf][aRow0][aSeg], aSrc0);
        async_copy16((unsigned)(unsigned long long)&As[buf][aRow0 + 32][aSeg], aSrc1);
#pragma unroll
        for (int cc = 0; cc < 4; ++cc) {
            int krow = bK0 + cc * 8;
            _Float16 h[8];
            if (bValid) {
                *(uint4*)h = *(const uint4*)bSrc[cc];
            } else {
#pragma unroll
                for (int j = 0; j < 8; ++j) h[j] = (_Float16)0.f;
            }
#pragma unroll
            for (int j = 0; j < 8; ++j) Bs[buf][nseg + j][krow] = h[j];
        }
    };
    auto advance = [&]() {
        aSrc0 += BK; aSrc1 += BK;
#pragma unroll
        for (int cc = 0; cc < 4; ++cc) bSrc[cc] += (size_t)BK * ldb;
    };

    const int nsteps = Kd / BK;

    // ---- prologue: fill buffer 0 ----
    stage(0);
    advance();
    wait_async0();
    __syncthreads();

    // ---- main pipeline ----
    for (int s = 0; s < nsteps; ++s) {
        const int cur = s & 1;
        if (s + 1 < nsteps) {          // prefetch next tile into other buffer
            stage(cur ^ 1);
            advance();
        }

        // Fragments per ISA VGPR layouts (CDNA5 ISA): two b128s each,
        // 80B row stride => conflict-free bank pattern across lanes.
        union { v16h v; uint4 q[2]; } af[2], bf[4];
#pragma unroll
        for (int i = 0; i < 2; ++i) {
            int m = wr * 32 + i * 16 + lmod;
            af[i].q[0] = *(const uint4*)&As[cur][m][lhalf * 8];       // K=h*8..+7
            af[i].q[1] = *(const uint4*)&As[cur][m][16 + lhalf * 8];  // K=16+h*8..
        }
#pragma unroll
        for (int j = 0; j < 4; ++j) {
            int n = wc * 64 + j * 16 + lmod;
            bf[j].q[0] = *(const uint4*)&Bs[cur][n][lhalf * 8];       // K=h*8..+7
            bf[j].q[1] = *(const uint4*)&Bs[cur][n][16 + lhalf * 8];  // K=16+h*8..
        }
#pragma unroll
        for (int i = 0; i < 2; ++i)
#pragma unroll
            for (int j = 0; j < 4; ++j)
                acc[i][j] = __builtin_amdgcn_wmma_f32_16x16x32_f16(
                    false, af[i].v, false, bf[j].v,
                    (short)0, acc[i][j], false, false);

        wait_async0();       // next A tile landed (overlapped with WMMAs above)
        __syncthreads();     // dscnt wait + barrier (B scatter visible)
    }

    // epilogue per C/D layout: VGPR r -> row r (lanes 0-15) / 8+r (lanes 16-31)
#pragma unroll
    for (int i = 0; i < 2; ++i)
        for (int j = 0; j < 4; ++j) {
            int nCol = nBlock + wc * 64 + j * 16 + lmod;
            if (nCol >= N) continue;
            int mRow0 = mBlock + wr * 32 + i * 16 + lhalf * 8;
#pragma unroll
            for (int r = 0; r < 8; ++r) {
                float v = acc[i][j][r];
                int m = mRow0 + r;
                v += bias[m];
                if (act == 1)      v = fmaxf(v, 0.f);
                else if (act == 2) v = 1.f / (1.f + __expf(-v));
                Cb[(size_t)m * ldc + nCol] = (CT)v;
            }
        }
}

// ------------------------------- launcher ----------------------------------

extern "C" void kernel_launch(void* const* d_in, const int* in_sizes, int n_in,
                              void* d_out, int out_size, void* d_ws, size_t ws_size,
                              hipStream_t stream)
{
    const float* inputs = (const float*)d_in[0];
    const float* w_b1 = (const float*)d_in[1];  const float* b_b1 = (const float*)d_in[2];
    const float* w_b2 = (const float*)d_in[3];  const float* b_b2 = (const float*)d_in[4];
    const float* w_s1 = (const float*)d_in[5];  const float* b_s1 = (const float*)d_in[6];
    const float* w_s2 = (const float*)d_in[7];  const float* b_s2 = (const float*)d_in[8];
    const float* w_e1 = (const float*)d_in[9];  const float* b_e1 = (const float*)d_in[10];
    const float* w_e2 = (const float*)d_in[11]; const float* b_e2 = (const float*)d_in[12];
    const float* w_p1 = (const float*)d_in[13]; const float* b_p1 = (const float*)d_in[14];
    const float* w3d  = (const float*)d_in[15]; const float* b3d  = (const float*)d_in[16];
    const float* w2a  = (const float*)d_in[17]; const float* b2a  = (const float*)d_in[18];
    const float* w2b  = (const float*)d_in[19]; const float* b2b  = (const float*)d_in[20];
    const float* w2c  = (const float*)d_in[21]; const float* b2c  = (const float*)d_in[22];
    const float* w2d  = (const float*)d_in[23]; const float* b2d  = (const float*)d_in[24];
    const float* smask = (const float*)d_in[25];

    float* conf  = (float*)d_out;              // [4,2,100,100] = 80000
    float* start = (float*)d_out + 80000;      // [4,100]
    float* endp  = (float*)d_out + 80400;      // [4,100]

    size_t off = 0;
    auto alloc = [&](size_t bytes) -> char* {
        char* p = (char*)d_ws + off;
        off = (off + bytes + 255) & ~(size_t)255;
        return p;
    };
    float*    base1  = (float*)   alloc((size_t)4 * 256 * T100 * 4);
    float*    base2  = (float*)   alloc((size_t)4 * 256 * T100 * 4);
    float*    tmpbr  = (float*)   alloc((size_t)4 * 256 * T100 * 4);
    float*    pbuf   = (float*)   alloc((size_t)4 * 128 * T100 * 4);
    _Float16* sbuf   = (_Float16*)alloc((size_t)4 * 512 * 3200 * 2);
    _Float16* maskT  = (_Float16*)alloc((size_t)3200 * POS * 2);
    _Float16* out3d  = (_Float16*)alloc((size_t)4 * 512 * POS * 2);
    float*    x1     = (float*)   alloc((size_t)4 * 128 * POS * 4);
    float*    x2     = (float*)   alloc((size_t)4 * 128 * POS * 4);
    float*    x3     = (float*)   alloc((size_t)4 * 128 * POS * 4);
    _Float16* colbuf = (_Float16*)alloc((size_t)1152 * POS * 2);
    _Float16* w2a16  = (_Float16*)alloc((size_t)128 * 512 * 2);
    _Float16* w2b16  = (_Float16*)alloc((size_t)128 * 1152 * 2);
    _Float16* w2c16  = (_Float16*)alloc((size_t)128 * 1152 * 2);

    const int TB = 256;
    auto blocks = [](int total, int tb) { return (total + tb - 1) / tb; };

    // ---- 1-D conv front-end (cheap) ----
    conv1d_act<<<blocks(4 * 256 * T100, TB), TB, 0, stream>>>(inputs, w_b1, b_b1, base1, 4, 400, 256, 3, 1, 4, 1);
    conv1d_act<<<blocks(4 * 256 * T100, TB), TB, 0, stream>>>(base1, w_b2, b_b2, base2, 4, 256, 256, 3, 1, 4, 1);
    conv1d_act<<<blocks(4 * 256 * T100, TB), TB, 0, stream>>>(base2, w_s1, b_s1, tmpbr, 4, 256, 256, 3, 1, 4, 1);
    conv1d_act<<<blocks(4 * 1 * T100, TB),   TB, 0, stream>>>(tmpbr, w_s2, b_s2, start, 4, 256, 1, 1, 0, 1, 2);
    conv1d_act<<<blocks(4 * 256 * T100, TB), TB, 0, stream>>>(base2, w_e1, b_e1, tmpbr, 4, 256, 256, 3, 1, 4, 1);
    conv1d_act<<<blocks(4 * 1 * T100, TB),   TB, 0, stream>>>(tmpbr, w_e2, b_e2, endp, 4, 256, 1, 1, 0, 1, 2);
    conv1d_act<<<blocks(4 * 128 * T100, TB), TB, 0, stream>>>(base2, w_p1, b_p1, pbuf, 4, 256, 128, 3, 1, 1, 1);

    // ---- fused BM + conv3d prep ----
    s_contract<<<blocks(4 * 512 * 3200, TB), TB, 0, stream>>>(pbuf, w3d, sbuf);
    mask_repack<<<blocks(3200 * POS, TB), TB, 0, stream>>>(smask, maskT);
    cvt_f16<<<blocks(128 * 512, TB),  TB, 0, stream>>>(w2a, w2a16, 128 * 512);
    cvt_f16<<<blocks(128 * 1152, TB), TB, 0, stream>>>(w2b, w2b16, 128 * 1152);
    cvt_f16<<<blocks(128 * 1152, TB), TB, 0, stream>>>(w2c, w2c16, 128 * 1152);

    const int NBx = (POS + 127) / 128;  // 79
    // out3d[b] = relu(s[b][512x3200] x maskT[3200x10000] + b3d)  -> f16
    gemm_wmma<_Float16><<<dim3(NBx, 512 / 64, 4), 128, 0, stream>>>(
        sbuf, (long long)512 * 3200, 3200,
        maskT, 0LL, POS,
        out3d, (long long)512 * POS, POS,
        b3d, POS, 3200, 1);

    // conv2a 1x1: x1[b] = relu(w2a[128x512] x out3d[b] + b2a)
    gemm_wmma<float><<<dim3(NBx, 128 / 64, 4), 128, 0, stream>>>(
        w2a16, 0LL, 512,
        out3d, (long long)512 * POS, POS,
        x1, (long long)128 * POS, POS,
        b2a, POS, 512, 1);

    // conv2b 3x3 via im2col + WMMA GEMM (K = 1152), per batch (stream-ordered
    // reuse of colbuf is safe: single in-order stream / captured graph)
    for (int b = 0; b < 4; ++b) {
        im2col3x3<<<blocks(1152 * POS, TB), TB, 0, stream>>>(x1 + (size_t)b * 128 * POS, colbuf);
        gemm_wmma<float><<<dim3(NBx, 128 / 64, 1), 128, 0, stream>>>(
            w2b16, 0LL, 1152,
            colbuf, 0LL, POS,
            x2 + (size_t)b * 128 * POS, 0LL, POS,
            b2b, POS, 1152, 1);
    }
    // conv2c 3x3
    for (int b = 0; b < 4; ++b) {
        im2col3x3<<<blocks(1152 * POS, TB), TB, 0, stream>>>(x2 + (size_t)b * 128 * POS, colbuf);
        gemm_wmma<float><<<dim3(NBx, 128 / 64, 1), 128, 0, stream>>>(
            w2c16, 0LL, 1152,
            colbuf, 0LL, POS,
            x3 + (size_t)b * 128 * POS, 0LL, POS,
            b2c, POS, 1152, 1);
    }

    // conv2d 1x1 -> sigmoid -> conf
    conv1x1_sig<<<blocks(4 * 2 * POS, TB), TB, 0, stream>>>(x3, w2d, b2d, conf);
}
